// FastQuantizedModel_82489141887435
// MI455X (gfx1250) — 8-bit matrix operand order repaired and checked by reading, compiled, not run on hardware
//
#include <hip/hip_runtime.h>
#include <stdint.h>


#define D_DIM 4096
#define N_TOK 4096

typedef __attribute__((ext_vector_type(8))) int v8i;

union Frag8 {
    v8i   v;
    uint2 u2[4];
    uint4 u4[2];
};

__device__ __forceinline__ int f2ord(float f) {
    int i = __float_as_int(f);
    return (i >= 0) ? i : (i ^ 0x7fffffff);
}
__device__ __forceinline__ float ord2f(int i) {
    return __int_as_float((i >= 0) ? i : (i ^ 0x7fffffff));
}
__device__ __forceinline__ int iclamp(int v, int lo, int hi) {
    return v < lo ? lo : (v > hi ? hi : v);
}

__device__ __forceinline__ uint32_t lds_off(const void* p) {
    return (uint32_t)(uintptr_t)p;
}
__device__ __forceinline__ void async_ld_b128(uint32_t lds_addr, const void* gptr) {
    asm volatile("global_load_async_to_lds_b128 %0, %1, off"
                 :: "v"(lds_addr), "v"(gptr)
                 : "memory");
}
__device__ __forceinline__ void wait_asynccnt_4() {
    asm volatile("s_wait_asynccnt 0x4" ::: "memory");
}
__device__ __forceinline__ void wait_asynccnt_0() {
    asm volatile("s_wait_asynccnt 0x0" ::: "memory");
}

__global__ void k_init_minmax(int* mm) {
    int t = threadIdx.x;
    if (t < 8) mm[t] = (t & 1) ? INT_MIN : INT_MAX;
}

__global__ __launch_bounds__(256) void k_minmax(const float* __restrict__ src,
                                                long long n, int* __restrict__ mm,
                                                int slot) {
    float lo = 3.402823466e38f, hi = -3.402823466e38f;
    for (long long i = (long long)blockIdx.x * blockDim.x + threadIdx.x; i < n;
         i += (long long)gridDim.x * blockDim.x) {
        float v = src[i];
        lo = fminf(lo, v);
        hi = fmaxf(hi, v);
    }
    __shared__ float slo[256];
    __shared__ float shi[256];
    int t = threadIdx.x;
    slo[t] = lo;
    shi[t] = hi;
    __syncthreads();
    for (int s = 128; s > 0; s >>= 1) {
        if (t < s) {
            slo[t] = fminf(slo[t], slo[t + s]);
            shi[t] = fmaxf(shi[t], shi[t + s]);
        }
        __syncthreads();
    }
    if (t == 0) {
        atomicMin(&mm[slot], f2ord(slo[0]));
        atomicMax(&mm[slot + 1], f2ord(shi[0]));
    }
}

__global__ __launch_bounds__(256) void k_quant_act(const float* __restrict__ x,
                                                   const int* __restrict__ mm, int slot,
                                                   uint8_t* __restrict__ xq,
                                                   long long n) {
    const float mn = ord2f(mm[slot]);
    const float mx = ord2f(mm[slot + 1]);
    const float inv = 127.0f / (mx - mn);
    for (long long i = (long long)blockIdx.x * blockDim.x + threadIdx.x; i < n;
         i += (long long)gridDim.x * blockDim.x) {
        int q = (int)rintf((x[i] - mn) * inv);
        xq[i] = (uint8_t)iclamp(q, 0, 127);
    }
}

__global__ __launch_bounds__(256) void k_quant_weight_T(const float* __restrict__ w,
                                                        const int* __restrict__ mm,
                                                        int slot,
                                                        int8_t* __restrict__ qT) {
    __shared__ int8_t tile[32][33];
    const float mn = ord2f(mm[slot]);
    const float mx = ord2f(mm[slot + 1]);
    const float inv = 15.0f / (mx - mn);
    const int n0 = blockIdx.x * 32;
    const int k0 = blockIdx.y * 32;
    const int tx = threadIdx.x & 31;
    const int ty = threadIdx.x >> 5;
#pragma unroll
    for (int j = 0; j < 4; ++j) {
        const int k = k0 + ty + j * 8;
        const float v = w[(long long)k * D_DIM + n0 + tx];
        const int q = iclamp((int)rintf((v - mn) * inv), 0, 15);
        tile[ty + j * 8][tx] = (int8_t)(q - 8);
    }
    __syncthreads();
#pragma unroll
    for (int j = 0; j < 4; ++j) {
        const int n = n0 + ty + j * 8;
        qT[(long long)n * D_DIM + k0 + tx] = tile[tx][ty + j * 8];
    }
}

#define LDA 80
#define LDB 80
#define LA_BYTES (128 * LDA)
#define LB_BYTES (128 * LDB)

__global__ __launch_bounds__(256, 2) void k_gemm_iu8(const uint8_t* __restrict__ A,
                                                     const int8_t* __restrict__ BT,
                                                     float* __restrict__ C,
                                                     const int* __restrict__ mm,
                                                     int slot) {
    __shared__ __align__(16) uint8_t la[2][LA_BYTES];
    __shared__ __align__(16) int8_t  lb[2][LB_BYTES];

    const int t    = threadIdx.x;
    const int wv   = t >> 5;
    const int lane = t & 31;
    const int wvm  = wv >> 1;
    const int wvn  = wv & 1;
    const int m0 = blockIdx.x * 128;
    const int n0 = blockIdx.y * 128;

    const int rw0 = t >> 2,         ko0 = (t & 3) << 4;
    const int rw1 = (t + 256) >> 2, ko1 = ko0;

    const uint8_t* gA0 = A + (long long)(m0 + rw0) * D_DIM + ko0;
    const uint8_t* gA1 = A + (long long)(m0 + rw1) * D_DIM + ko1;
    const int8_t*  gB0 = BT + (long long)(n0 + rw0) * D_DIM + ko0;
    const int8_t*  gB1 = BT + (long long)(n0 + rw1) * D_DIM + ko1;

    const uint32_t lA0 = lds_off(&la[0][rw0 * LDA + ko0]);
    const uint32_t lA1 = lds_off(&la[0][rw1 * LDA + ko1]);
    const uint32_t lB0 = lds_off(&lb[0][rw0 * LDB + ko0]);
    const uint32_t lB1 = lds_off(&lb[0][rw1 * LDB + ko1]);

    v8i acc[8] = {v8i{}, v8i{}, v8i{}, v8i{}, v8i{}, v8i{}, v8i{}, v8i{}};

    const int arow0 = wvm * 32 + (lane & 15);
    const int abase = (lane >> 4) << 4;
    const int bcol0 = wvn * 64 + (lane & 15);
    const int kb    = (lane >> 4) << 4;

    async_ld_b128(lA0, gA0);
    async_ld_b128(lA1, gA1);
    async_ld_b128(lB0, gB0);
    async_ld_b128(lB1, gB1);

    const int NSTEP = D_DIM / 64;
#pragma clang loop unroll(disable)
    for (int i = 0; i < NSTEP; ++i) {
        const int buf = i & 1;
        gA0 += 64; gA1 += 64; gB0 += 64; gB1 += 64;
        if (i + 1 < NSTEP) {
            const uint32_t bo = (uint32_t)((i + 1) & 1) * LA_BYTES;
            async_ld_b128(lA0 + bo, gA0);
            async_ld_b128(lA1 + bo, gA1);
            async_ld_b128(lB0 + bo, gB0);
            async_ld_b128(lB1 + bo, gB1);
            wait_asynccnt_4();
        } else {
            wait_asynccnt_0();
        }
        __syncthreads();

        Frag8 af[2];
#pragma unroll
        for (int f = 0; f < 2; ++f) {
            const uint8_t* ap = &la[buf][(arow0 + f * 16) * LDA + abase];
            af[f].u2[0] = *(const uint2*)(ap);
            af[f].u2[1] = *(const uint2*)(ap + 8);
            af[f].u2[2] = *(const uint2*)(ap + 32);
            af[f].u2[3] = *(const uint2*)(ap + 40);
        }
        Frag8 bf[4];
#pragma unroll
        for (int j = 0; j < 4; ++j) {
            const int8_t* bp = &lb[buf][(bcol0 + j * 16) * LDB + kb];
            bf[j].u4[0] = *(const uint4*)(bp);
            bf[j].u4[1] = *(const uint4*)(bp + 32);
        }
#pragma unroll
        for (int j = 0; j < 4; ++j) {
#pragma unroll
            for (int f = 0; f < 2; ++f) {
                acc[f * 4 + j] = __builtin_amdgcn_wmma_i32_16x16x64_iu8(
                    false, af[f].v, true, bf[j].v, acc[f * 4 + j], false, false);
            }
        }
        __syncthreads();
    }

    const float wmn   = ord2f(mm[slot]);
    const float wmx   = ord2f(mm[slot + 1]);
    const float scale = (wmx - wmn) / 15.0f;

    const int mbase = m0 + wvm * 32 + ((lane >> 4) << 3);
    const int nbase = n0 + wvn * 64 + (lane & 15);
#pragma unroll
    for (int f = 0; f < 2; ++f) {
#pragma unroll
        for (int j = 0; j < 4; ++j) {
#pragma unroll
            for (int r = 0; r < 8; ++r) {
                C[(long long)(mbase + f * 16 + r) * D_DIM + nbase + j * 16] =
                    (float)acc[f * 4 + j][r] * scale + wmn;
            }
        }
    }
}

extern "C" void kernel_launch(void* const* d_in, const int* in_sizes, int n_in,
                              void* d_out, int out_size, void* d_ws, size_t ws_size,
                              hipStream_t stream) {
    (void)in_sizes; (void)n_in; (void)out_size; (void)ws_size;
    const float* x  = (const float*)d_in[0];
    const float* w1 = (const float*)d_in[1];
    const float* w2 = (const float*)d_in[2];
    float* out = (float*)d_out;

    const long long ND = (long long)N_TOK * D_DIM;

    uint8_t* ws8 = (uint8_t*)d_ws;
    int*     mm  = (int*)ws8;
    uint8_t* xq  = ws8 + 256;
    uint8_t* hq  = xq + ND;
    int8_t*  q1T = (int8_t*)(hq + ND);
    int8_t*  q2T = q1T + ND;

    k_init_minmax<<<1, 32, 0, stream>>>(mm);

    k_minmax<<<1024, 256, 0, stream>>>(x,  ND, mm, 0);
    k_minmax<<<1024, 256, 0, stream>>>(w1, ND, mm, 2);
    k_minmax<<<1024, 256, 0, stream>>>(w2, ND, mm, 6);

    k_quant_act<<<2048, 256, 0, stream>>>(x, mm, 0, xq, ND);
    dim3 gt(D_DIM / 32, D_DIM / 32);
    k_quant_weight_T<<<gt, 256, 0, stream>>>(w1, mm, 2, q1T);
    k_quant_weight_T<<<gt, 256, 0, stream>>>(w2, mm, 6, q2T);

    dim3 gg(N_TOK / 128, D_DIM / 128);
    k_gemm_iu8<<<gg, 256, 0, stream>>>(xq, q1T, out, mm, 2);

    k_minmax<<<1024, 256, 0, stream>>>(out, ND, mm, 4);
    k_quant_act<<<2048, 256, 0, stream>>>(out, mm, 4, hq, ND);

    k_gemm_iu8<<<gg, 256, 0, stream>>>(hq, q2T, out, mm, 6);
}
